// MazeGNN_37349035606215
// MI455X (gfx1250) — matrix operand order repaired and checked by reading, not compiled, not run on hardware
//
#include <hip/hip_runtime.h>
#include <math.h>
#include <stdint.h>

typedef __attribute__((ext_vector_type(16))) _Float16 v16h;
typedef __attribute__((ext_vector_type(8)))  _Float16 v8h;
typedef __attribute__((ext_vector_type(8)))  float    v8f;
typedef __attribute__((ext_vector_type(4)))  float    v4f;

#define WAVES 8
#define TPB   (WAVES * 32)
#define SA_STRIDE 136   // 128 halves + 8 pad (keeps 16B alignment, spreads banks)
#define SI_STRIDE 72    // 64 halves + 8 pad

// ---------------------------------------------------------------------------
// WMMA fragment loaders
// A fragment (16x32 f16, M x K): per ISA layout
//   lanes 0-15  (M = lane):    halves[0..7] = K 0..7,  halves[8..15] = K 16..23
//   lanes 16-31 (M = lane-16): halves[0..7] = K 8..15, halves[8..15] = K 24..31
// ---------------------------------------------------------------------------
__device__ __forceinline__ v16h load_a_frag(const _Float16* sA, int stride,
                                            int lane, int ks) {
    const int row   = lane & 15;
    const int kbase = ks * 32 + ((lane >> 4) << 3);
    const _Float16* p = sA + row * stride + kbase;
    v8h lo = *(const v8h*)(p);
    v8h hi = *(const v8h*)(p + 16);
    v16h a;
#pragma unroll
    for (int i = 0; i < 8; ++i) { a[i] = lo[i]; a[i + 8] = hi[i]; }
    return a;
}

// B fragment: weights are prepacked so each lane's 16 halves are contiguous.
__device__ __forceinline__ v16h load_b_frag(const _Float16* wp, int ksteps,
                                            int t, int ks, int lane) {
    const _Float16* p = wp + (size_t)(((t * ksteps) + ks) * 32 + lane) * 16;
    return *(const v16h*)p;   // 32B aligned
}

// ---------------------------------------------------------------------------
// 16-row x (K1 -> 64 -> 64) two-layer MLP on one wave.
// Layer1: relu(A[16xK1] @ W1 + b1) staged to LDS (sI).
// Layer2: acc2[t] = Inter @ W2 + b2 (no relu), t = 0..3 N-tiles of 16.
// C layout: lane holds col=(lane&15) of rows rbase..rbase+7, rbase=(lane>>4)*8.
// ---------------------------------------------------------------------------
__device__ __forceinline__ void mlp2_tile(const _Float16* sA, int strideA, int K1,
                                          _Float16* sI, int strideI,
                                          const _Float16* w1p, const float* b1,
                                          const _Float16* w2p, const float* b2,
                                          int lane, v8f* acc2) {
    const int ks1   = K1 >> 5;
    const int col   = lane & 15;
    const int rbase = (lane >> 4) * 8;

#pragma unroll
    for (int t = 0; t < 4; ++t) {
        float bv = b1[t * 16 + col];
        v8f acc;
#pragma unroll
        for (int v = 0; v < 8; ++v) acc[v] = bv;
        for (int ks = 0; ks < ks1; ++ks) {
            v16h a = load_a_frag(sA, strideA, lane, ks);
            v16h b = load_b_frag(w1p, ks1, t, ks, lane);
            acc = __builtin_amdgcn_wmma_f32_16x16x32_f16(
                false, a, false, b, (short)0, acc, false, false);
        }
#pragma unroll
        for (int v = 0; v < 8; ++v)
            sI[(rbase + v) * strideI + t * 16 + col] =
                (_Float16)fmaxf(acc[v], 0.0f);
    }
    // layer 2: K = 64 (2 k-steps); within-wave DS ops are in-order.
#pragma unroll
    for (int t = 0; t < 4; ++t) {
        float bv = b2[t * 16 + col];
        v8f acc;
#pragma unroll
        for (int v = 0; v < 8; ++v) acc[v] = bv;
#pragma unroll
        for (int ks = 0; ks < 2; ++ks) {
            v16h a = load_a_frag(sI, strideI, lane, ks);
            v16h b = load_b_frag(w2p, 2, t, ks, lane);
            acc = __builtin_amdgcn_wmma_f32_16x16x32_f16(
                false, a, false, b, (short)0, acc, false, false);
        }
        acc2[t] = acc;
    }
}

// ---------------------------------------------------------------------------
// Weight prepack: f32 [K x 64] -> f16 fragment order.
// dst[((t*ksteps+s)*32 + l)*16 + j] = W[s*32 + (l>>4)*8 + (j&7) + (j>>3)*16][t*16 + (l&15)]
// ---------------------------------------------------------------------------
__global__ void prepack_w(const float* __restrict__ src,
                          _Float16* __restrict__ dst, int K) {
    int tid = blockIdx.x * blockDim.x + threadIdx.x;
    if (tid >= K * 64) return;
    int j = tid & 15;
    int l = (tid >> 4) & 31;
    int rest = tid >> 9;
    int ksteps = K >> 5;
    int s = rest % ksteps;
    int t = rest / ksteps;
    int krow = s * 32 + ((l >> 4) << 3) + (j & 7) + ((j >> 3) << 4);
    int colw = t * 16 + (l & 15);
    dst[tid] = (_Float16)src[krow * 64 + colw];
}

__global__ void zero_kernel(float* __restrict__ p, int n) {
    int i = blockIdx.x * blockDim.x + threadIdx.x;
    if (i < n) p[i] = 0.0f;
}

// ---------------------------------------------------------------------------
// Encoder: h = relu(mlp2(x)) + (x @ ip_w + ip_b); enc = x @ ip_w + ip_b
// ---------------------------------------------------------------------------
__global__ void encoder_kernel(const float* __restrict__ x,
                               const float* __restrict__ ew1, const float* __restrict__ eb1,
                               const float* __restrict__ ew2, const float* __restrict__ eb2,
                               const float* __restrict__ ipw, const float* __restrict__ ipb,
                               _Float16* __restrict__ h, _Float16* __restrict__ enc,
                               int N) {
    int idx = blockIdx.x * blockDim.x + threadIdx.x;
    if (idx >= N * 64) return;
    int n = idx >> 6, f = idx & 63;
    float x0 = x[n * 2], x1 = x[n * 2 + 1];
    float ie = x0 * ipw[f] + x1 * ipw[64 + f] + ipb[f];
    float acc = eb2[f];
#pragma unroll 8
    for (int j = 0; j < 32; ++j) {
        float hj = fmaxf(x0 * ew1[j] + x1 * ew1[32 + j] + eb1[j], 0.0f);
        acc += hj * ew2[j * 64 + f];
    }
    acc = fmaxf(acc, 0.0f) + ie;
    h[idx]   = (_Float16)acc;
    enc[idx] = (_Float16)ie;
}

// ---------------------------------------------------------------------------
// Message kernel: per wave, 16 edges. A = [h[dst] | h[src]] (16 x 128),
// two-layer MLP, atomic scatter-add into agg[dst].
// ---------------------------------------------------------------------------
__global__ void __launch_bounds__(TPB) message_kernel(
    const _Float16* __restrict__ h, const int* __restrict__ ei,
    const _Float16* __restrict__ w1p, const float* __restrict__ b1,
    const _Float16* __restrict__ w2p, const float* __restrict__ b2,
    float* __restrict__ agg, int E) {
    __shared__ __align__(32) _Float16 sA[WAVES][16 * SA_STRIDE];
    __shared__ __align__(32) _Float16 sI[WAVES][16 * SI_STRIDE];
    __shared__ int sNode[WAVES][16];

    const int lane = threadIdx.x & 31;
    const int w    = threadIdx.x >> 5;
    const int eBase = (blockIdx.x * WAVES + w) * 16;
    const int* srcI = ei;
    const int* dstI = ei + E;

    {   // stage: lane (row, half): half 0 -> h[dst] (cols 0..63), half 1 -> h[src]
        int row  = lane & 15;
        int half = lane >> 4;
        int e    = eBase + row;
        int ec   = e < E ? e : E - 1;
        int node = half ? srcI[ec] : dstI[ec];
        const v8h* sp = (const v8h*)(h + (size_t)node * 64);
        v8h* dp = (v8h*)(&sA[w][row * SA_STRIDE + half * 64]);
#pragma unroll
        for (int i = 0; i < 8; ++i) dp[i] = sp[i];
        if (!half) sNode[w][row] = node;
    }

    v8f acc2[4];
    mlp2_tile(sA[w], SA_STRIDE, 128, sI[w], SI_STRIDE, w1p, b1, w2p, b2, lane, acc2);

    const int col = lane & 15, rbase = (lane >> 4) * 8;
#pragma unroll
    for (int t = 0; t < 4; ++t) {
        int c = t * 16 + col;
#pragma unroll
        for (int v = 0; v < 8; ++v) {
            int row = rbase + v;
            int e   = eBase + row;
            if (e < E)
                atomicAdd(&agg[(size_t)sNode[w][row] * 64 + c], acc2[t][v]);
        }
    }
}

// ---------------------------------------------------------------------------
// Concat MLP: h = relu(relu([h | enc] @ cc_w1 + b1) @ cc_w2 + b2)
// ---------------------------------------------------------------------------
__global__ void __launch_bounds__(TPB) concat_kernel(
    _Float16* __restrict__ h, const _Float16* __restrict__ enc,
    const _Float16* __restrict__ w1p, const float* __restrict__ b1,
    const _Float16* __restrict__ w2p, const float* __restrict__ b2, int N) {
    __shared__ __align__(32) _Float16 sA[WAVES][16 * SA_STRIDE];
    __shared__ __align__(32) _Float16 sI[WAVES][16 * SI_STRIDE];

    const int lane = threadIdx.x & 31;
    const int w    = threadIdx.x >> 5;
    const int nBase = (blockIdx.x * WAVES + w) * 16;

    {
        int row  = lane & 15;
        int half = lane >> 4;
        int n    = nBase + row;
        int nc   = n < N ? n : N - 1;
        const _Float16* srcp = half ? enc : h;
        const v8h* sp = (const v8h*)(srcp + (size_t)nc * 64);
        v8h* dp = (v8h*)(&sA[w][row * SA_STRIDE + half * 64]);
#pragma unroll
        for (int i = 0; i < 8; ++i) dp[i] = sp[i];
    }

    v8f acc2[4];
    mlp2_tile(sA[w], SA_STRIDE, 128, sI[w], SI_STRIDE, w1p, b1, w2p, b2, lane, acc2);

    const int col = lane & 15, rbase = (lane >> 4) * 8;
#pragma unroll
    for (int t = 0; t < 4; ++t) {
#pragma unroll
        for (int v = 0; v < 8; ++v) {
            int n = nBase + rbase + v;
            if (n < N)
                h[(size_t)n * 64 + t * 16 + col] = (_Float16)fmaxf(acc2[t][v], 0.0f);
        }
    }
}

// ---------------------------------------------------------------------------
// Update MLP: h = relu(([h | agg] @ upd_w1 ... @ upd_w2 + b2) + h)
// ---------------------------------------------------------------------------
__global__ void __launch_bounds__(TPB) update_kernel(
    _Float16* __restrict__ h, const float* __restrict__ agg,
    const _Float16* __restrict__ w1p, const float* __restrict__ b1,
    const _Float16* __restrict__ w2p, const float* __restrict__ b2, int N) {
    __shared__ __align__(32) _Float16 sA[WAVES][16 * SA_STRIDE];
    __shared__ __align__(32) _Float16 sI[WAVES][16 * SI_STRIDE];

    const int lane = threadIdx.x & 31;
    const int w    = threadIdx.x >> 5;
    const int nBase = (blockIdx.x * WAVES + w) * 16;

    {
        int row  = lane & 15;
        int half = lane >> 4;
        int n    = nBase + row;
        int nc   = n < N ? n : N - 1;
        if (!half) {
            const v8h* sp = (const v8h*)(h + (size_t)nc * 64);
            v8h* dp = (v8h*)(&sA[w][row * SA_STRIDE]);
#pragma unroll
            for (int i = 0; i < 8; ++i) dp[i] = sp[i];
        } else {
            const v4f* sp = (const v4f*)(agg + (size_t)nc * 64);
            _Float16* dp = &sA[w][row * SA_STRIDE + 64];
#pragma unroll
            for (int i = 0; i < 16; ++i) {
                v4f t4 = sp[i];
#pragma unroll
                for (int j = 0; j < 4; ++j) dp[i * 4 + j] = (_Float16)t4[j];
            }
        }
    }

    v8f acc2[4];
    mlp2_tile(sA[w], SA_STRIDE, 128, sI[w], SI_STRIDE, w1p, b1, w2p, b2, lane, acc2);

    const int col = lane & 15, rbase = (lane >> 4) * 8;
#pragma unroll
    for (int t = 0; t < 4; ++t) {
#pragma unroll
        for (int v = 0; v < 8; ++v) {
            int row = rbase + v;
            int n = nBase + row;
            if (n < N) {
                float skip = (float)sA[w][row * SA_STRIDE + t * 16 + col];
                h[(size_t)n * 64 + t * 16 + col] =
                    (_Float16)fmaxf(acc2[t][v] + skip, 0.0f);
            }
        }
    }
}

// ---------------------------------------------------------------------------
// Decoder + log_softmax (tiny: N x (64->64->2))
// ---------------------------------------------------------------------------
__global__ void decoder_kernel(const _Float16* __restrict__ h,
                               const float* __restrict__ w1, const float* __restrict__ b1,
                               const float* __restrict__ w2, const float* __restrict__ b2,
                               float* __restrict__ out, int N) {
    int n = blockIdx.x * blockDim.x + threadIdx.x;
    if (n >= N) return;
    float hr[64];
#pragma unroll
    for (int i = 0; i < 64; ++i) hr[i] = (float)h[(size_t)n * 64 + i];
    float o0 = b2[0], o1 = b2[1];
    for (int j = 0; j < 64; ++j) {
        float a = b1[j];
#pragma unroll
        for (int i = 0; i < 64; ++i) a += hr[i] * w1[i * 64 + j];
        a = fmaxf(a, 0.0f);
        o0 += a * w2[j * 2 + 0];
        o1 += a * w2[j * 2 + 1];
    }
    float m   = fmaxf(o0, o1);
    float lse = m + logf(expf(o0 - m) + expf(o1 - m));
    out[n * 2 + 0] = o0 - lse;
    out[n * 2 + 1] = o1 - lse;
}

// ---------------------------------------------------------------------------
extern "C" void kernel_launch(void* const* d_in, const int* in_sizes, int n_in,
                              void* d_out, int out_size, void* d_ws, size_t ws_size,
                              hipStream_t stream) {
    const float* x      = (const float*)d_in[0];
    const int*   ei     = (const int*)d_in[1];
    const float* enc_w1 = (const float*)d_in[3];
    const float* enc_b1 = (const float*)d_in[4];
    const float* enc_w2 = (const float*)d_in[5];
    const float* enc_b2 = (const float*)d_in[6];
    const float* dec_w1 = (const float*)d_in[7];
    const float* dec_b1 = (const float*)d_in[8];
    const float* dec_w2 = (const float*)d_in[9];
    const float* dec_b2 = (const float*)d_in[10];
    const float* ip_w   = (const float*)d_in[11];
    const float* ip_b   = (const float*)d_in[12];
    const float* cc_w1  = (const float*)d_in[13];
    const float* cc_b1  = (const float*)d_in[14];
    const float* cc_w2  = (const float*)d_in[15];
    const float* cc_b2  = (const float*)d_in[16];
    const float* msg_w1 = (const float*)d_in[17];
    const float* msg_b1 = (const float*)d_in[18];
    const float* msg_w2 = (const float*)d_in[19];
    const float* msg_b2 = (const float*)d_in[20];
    const float* upd_w1 = (const float*)d_in[21];
    const float* upd_b1 = (const float*)d_in[22];
    const float* upd_w2 = (const float*)d_in[23];
    const float* upd_b2 = (const float*)d_in[24];

    const int N = in_sizes[0] / 2;
    const int E = in_sizes[1] / 2;
    const int L = in_sizes[18] / 64;   // msg_b1 is [L,64]

    // eff = min(L, max(4, floor(log2(N))))
    int lg = 0; while ((1LL << (lg + 1)) <= (long long)N) ++lg;
    int mx = lg > 4 ? lg : 4;
    int eff = L < mx ? L : mx;

    // workspace carve-out (256B aligned)
    uint8_t* ws = (uint8_t*)d_ws;
    size_t off = 0;
    auto grab = [&](size_t bytes) -> uint8_t* {
        uint8_t* p = ws + off;
        off = (off + bytes + 255) & ~(size_t)255;
        return p;
    };
    _Float16* h    = (_Float16*)grab((size_t)N * 64 * 2);
    _Float16* enc  = (_Float16*)grab((size_t)N * 64 * 2);
    float*    agg  = (float*)   grab((size_t)N * 64 * 4);
    _Float16* wpck = (_Float16*)grab((12288 + (size_t)L * 24576) * 2);
    (void)ws_size; (void)n_in; (void)out_size;

    // 1) encoder + input_preserve
    encoder_kernel<<<(N * 64 + 255) / 256, 256, 0, stream>>>(
        x, enc_w1, enc_b1, enc_w2, enc_b2, ip_w, ip_b, h, enc, N);

    // 2) prepack weights into WMMA fragment order (f16)
    prepack_w<<<32, 256, 0, stream>>>(cc_w1, wpck, 128);
    prepack_w<<<16, 256, 0, stream>>>(cc_w2, wpck + 8192, 64);
    for (int i = 0; i < eff; ++i) {
        _Float16* base = wpck + 12288 + (size_t)i * 24576;
        prepack_w<<<32, 256, 0, stream>>>(msg_w1 + (size_t)i * 8192, base, 128);
        prepack_w<<<16, 256, 0, stream>>>(msg_w2 + (size_t)i * 4096, base + 8192, 64);
        prepack_w<<<32, 256, 0, stream>>>(upd_w1 + (size_t)i * 8192, base + 12288, 128);
        prepack_w<<<16, 256, 0, stream>>>(upd_w2 + (size_t)i * 4096, base + 20480, 64);
    }

    const int nodeBlocks = (N + 16 * WAVES - 1) / (16 * WAVES);
    const int edgeBlocks = (E + 16 * WAVES - 1) / (16 * WAVES);

    // 3) message-passing layers
    for (int i = 0; i < eff; ++i) {
        if (i > 0)
            concat_kernel<<<nodeBlocks, TPB, 0, stream>>>(
                h, enc, wpck, cc_b1, wpck + 8192, cc_b2, N);
        zero_kernel<<<(N * 64 + 255) / 256, 256, 0, stream>>>(agg, N * 64);
        _Float16* base = wpck + 12288 + (size_t)i * 24576;
        message_kernel<<<edgeBlocks, TPB, 0, stream>>>(
            h, ei, base, msg_b1 + i * 64, base + 8192, msg_b2 + i * 64, agg, E);
        update_kernel<<<nodeBlocks, TPB, 0, stream>>>(
            h, agg, base + 12288, upd_b1 + i * 64, base + 20480, upd_b2 + i * 64, N);
    }

    // 4) decoder + log_softmax
    decoder_kernel<<<(N + 255) / 256, 256, 0, stream>>>(
        h, dec_w1, dec_b1, dec_w2, dec_b2, (float*)d_out, N);
}
